// AttentionSTModule_42666205118808
// MI455X (gfx1250) — matrix operand order repaired and checked by reading, not compiled, not run on hardware
//
#include <hip/hip_runtime.h>

// ---------------------------------------------------------------------------
// AttentionSTModule for MI455X (gfx1250): bf16 WMMA pipeline + TDM staging.
// B=2, C=128, T=16, H=W=16, HEADS=8, DH=32, HID=256, MLP_H=512
// ---------------------------------------------------------------------------

typedef unsigned short u16;
typedef unsigned int   u32;
typedef unsigned long long u64;
typedef __attribute__((ext_vector_type(16))) __bf16 v16bf;
typedef __attribute__((ext_vector_type(8)))  float  v8f;
typedef __attribute__((ext_vector_type(4)))  unsigned int u32x4;
typedef __attribute__((ext_vector_type(8)))  int  i32x8;
typedef __attribute__((ext_vector_type(4)))  int  i32x4;

#define CB   128
#define TT   16
#define HW   256
#define NTOK 4096          // tokens per batch (H*W*T)
#define MTOK 8192          // total tokens (B * NTOK)
#define HEADS 8
#define DH   32
#define HID  256
#define MLPH 512
#define KF   160           // fusion K (129 padded to multiple of 32)
#define ATT_SCALE 0.17677669529663687f   // 32^-0.5

#if __has_builtin(__builtin_amdgcn_tensor_load_to_lds)
#define HAVE_TDM 1
#else
#define HAVE_TDM 0
#endif

union FragU {
  v16bf v;
  uint4 q[2];
  u16   u[16];
};

__device__ __forceinline__ u16 f2bf(float x) {
  u32 u = __float_as_uint(x);
  u32 r = u + 0x7FFFu + ((u >> 16) & 1u);   // round-to-nearest-even
  return (u16)(r >> 16);
}

__device__ __forceinline__ float gelu_f(float x) {
  return 0.5f * x * (1.0f + erff(x * 0.70710678118654752f));
}

__device__ __forceinline__ v8f wmma_bf16(v16bf a, v16bf b, v8f c) {
  return __builtin_amdgcn_wmma_f32_16x16x32_bf16(false, a, false, b, (short)0, c,
                                                 false, false);
}

// A-fragment: 16x32 bf16 (M x K), row-major A[M][lda].
__device__ __forceinline__ v16bf load_a_frag(const u16* __restrict__ A, int lda,
                                             int row0, int k0) {
  int lane = threadIdx.x & 31;
  int m = lane & 15, half = lane >> 4;
  const u16* p = A + (size_t)(row0 + m) * lda + k0;
  FragU f;
  f.q[0] = *(const uint4*)(p + 8 * half);
  f.q[1] = *(const uint4*)(p + 16 + 8 * half);
  return f.v;
}

// B-fragment: 32x16 bf16 (K x N), supplied as W^T row-major Wt[N][ldb(=K)].
__device__ __forceinline__ v16bf load_b_frag(const u16* __restrict__ Wt, int ldb,
                                             int n0, int k0) {
  int lane = threadIdx.x & 31;
  int n = lane & 15, half = lane >> 4;
  const u16* p = Wt + (size_t)(n0 + n) * ldb + k0;
  FragU f;
  f.q[0] = *(const uint4*)(p + 8 * half);
  f.q[1] = *(const uint4*)(p + 16 + 8 * half);
  return f.v;
}

__device__ __forceinline__ float red16_max(float v) {
  v = fmaxf(v, __shfl_xor(v, 1, 32));
  v = fmaxf(v, __shfl_xor(v, 2, 32));
  v = fmaxf(v, __shfl_xor(v, 4, 32));
  v = fmaxf(v, __shfl_xor(v, 8, 32));
  return v;
}
__device__ __forceinline__ float red16_sum(float v) {
  v += __shfl_xor(v, 1, 32);
  v += __shfl_xor(v, 2, 32);
  v += __shfl_xor(v, 4, 32);
  v += __shfl_xor(v, 8, 32);
  return v;
}
__device__ __forceinline__ float red32_sum(float v) {
  v += __shfl_xor(v, 1, 32);
  v += __shfl_xor(v, 2, 32);
  v += __shfl_xor(v, 4, 32);
  v += __shfl_xor(v, 8, 32);
  v += __shfl_xor(v, 16, 32);
  return v;
}

#if HAVE_TDM
// TDM: stage a 32x32 bf16 tile (rows x rowlen, element stride between rows)
// from global into LDS (packed [32][32] row-major).  D# layout per CDNA5 ISA
// CDNA5 ISA: group0 = {flags, lds_addr, global_addr, type=2},
// group1 = {data_size=2B, tensor dims, tile dims, dim0 stride}, groups 2/3 = 0
// (2D tile: tile_dim2 = 0), workgroup_mask = 0 (not in a cluster).
// This toolchain exposes the 6-operand builtin:
//   (u32x4 g0, i32x8 g1, i32x4 g2, i32x4 g3, i32x8 extra, i32 cpol)
__device__ __forceinline__ void tdm_stage_32x32(const u16* g, u32 lds_off,
                                                u32 stride_elems) {
  u64 ga = (u64)(uintptr_t)g;
  u32x4 g0;
  g0[0] = 1u;                                        // count=1, user desc
  g0[1] = lds_off;                                   // lds_addr (bytes)
  g0[2] = (u32)ga;                                   // global_addr[31:0]
  g0[3] = ((u32)(ga >> 32) & 0x01FFFFFFu) | (2u << 30);  // addr[56:32] | type=2
  i32x8 g1;
  g1[0] = (int)(1u << 16);                           // data_size=1 -> 2 bytes
  g1[1] = (int)((32u & 0xFFFFu) << 16);              // tensor_dim0 lo16 -> [63:48]
  g1[2] = (int)((32u >> 16) | ((32u & 0xFFFFu) << 16)); // dim0 hi / dim1 lo
  g1[3] = (int)((32u >> 16) | (32u << 16));          // dim1 hi | tile_dim0=32
  g1[4] = (int)(32u);                                // tile_dim1=32, tile_dim2=0
  g1[5] = (int)stride_elems;                         // tensor_dim0_stride lo32
  g1[6] = 0;                                         // stride hi16 | dim1_stride lo
  g1[7] = 0;
  i32x4 gz4 = {0, 0, 0, 0};
  i32x8 gz8 = {0, 0, 0, 0, 0, 0, 0, 0};
  __builtin_amdgcn_tensor_load_to_lds(g0, g1, gz4, gz4, gz8, 0);
}
#endif

// ---------------------------------------------------------------------------
// Weight prep: f32 [K][N] -> bf16 W^T [N][Kp] (zero pad K..Kp)
// ---------------------------------------------------------------------------
__global__ void wprep_kernel(const float* __restrict__ W, u16* __restrict__ Wt,
                             int K, int N, int Kp, int total) {
  int i = blockIdx.x * blockDim.x + threadIdx.x;
  if (i >= total) return;
  int k = i % Kp;
  int n = i / Kp;
  Wt[i] = (k < K) ? f2bf(W[(size_t)k * N + n]) : (u16)0;
}

// ---------------------------------------------------------------------------
// Pack x (B,C,T,H,W) f32 + frame idx -> xf bf16 [MTOK][KF]
// ---------------------------------------------------------------------------
__global__ void pack_x_kernel(const float* __restrict__ x,
                              const float* __restrict__ fr,
                              u16* __restrict__ xf, int total) {
  int i = blockIdx.x * blockDim.x + threadIdx.x;
  if (i >= total) return;
  int f  = i % KF;
  int tk = i / KF;
  int p  = tk & 255;
  int bt = tk >> 8;
  int t  = bt & 15;
  int b  = bt >> 4;
  float val;
  if (f < CB)        val = x[(((size_t)(b * CB + f) * TT + t) << 8) + p];
  else if (f == CB)  val = fr[t];
  else               val = 0.0f;
  xf[(size_t)tk * KF + f] = f2bf(val);
}

// ---------------------------------------------------------------------------
// LayerNorm: one wave per token (128 features), f32 in -> bf16 out
// ---------------------------------------------------------------------------
__global__ __launch_bounds__(256) void ln_kernel(const float* __restrict__ xin,
                                                 const float* __restrict__ g,
                                                 const float* __restrict__ bta,
                                                 u16* __restrict__ xout, int ntok) {
  int wave = threadIdx.x >> 5;
  int lane = threadIdx.x & 31;
  int tok = blockIdx.x * (blockDim.x >> 5) + wave;
  if (tok >= ntok) return;
  const float* px = xin + (size_t)tok * CB;
  float v[4], s = 0.0f;
#pragma unroll
  for (int j = 0; j < 4; ++j) { v[j] = px[lane + 32 * j]; s += v[j]; }
  s = red32_sum(s);
  float mean = s * (1.0f / 128.0f);
  float var = 0.0f;
#pragma unroll
  for (int j = 0; j < 4; ++j) { float d = v[j] - mean; var += d * d; }
  var = red32_sum(var) * (1.0f / 128.0f);
  float rstd = rsqrtf(var + 1e-5f);
#pragma unroll
  for (int j = 0; j < 4; ++j) {
    int f = lane + 32 * j;
    float o = (v[j] - mean) * rstd * g[f] + bta[f];
    xout[(size_t)tok * CB + f] = f2bf(o);
  }
}

// ---------------------------------------------------------------------------
// WMMA GEMM: out = epilogue(A[M][lda] @ Wt^T + bias)
// block = 256 threads (8 waves); wave tile 16x64; block tile 64 M x 128 N
// MODE 0: GELU -> bf16 [M][N]
// MODE 1: bf16 [M][N]
// MODE 2: fusion scatter -> xs f32 [B][NTOK][C]
// MODE 3: V transpose scatter -> bf16 [B][HID][NTOK]
// MODE 4: f32 residual add -> [M][N] (N==128)
// MODE 5: final: resid + val -> d_out (b,c,t,h,w) f32
// ---------------------------------------------------------------------------
template <int MODE>
__global__ __launch_bounds__(256) void gemm_kernel(
    const u16* __restrict__ A, int lda, const u16* __restrict__ Bw, int ldb,
    const float* __restrict__ bias, int M, int N, int K,
    const float* __restrict__ resid, void* __restrict__ outp) {
  int lane = threadIdx.x & 31;
  int w = threadIdx.x >> 5;
  int wm = w & 3, wn = w >> 2;
  int row0  = blockIdx.y * 64 + wm * 16;
  int ncol0 = blockIdx.x * 128 + wn * 64;
  int half = lane >> 4, lidx = lane & 15;

  v8f acc[4] = {};
  for (int k0 = 0; k0 < K; k0 += 32) {
    v16bf af = load_a_frag(A, lda, row0, k0);
#pragma unroll
    for (int t = 0; t < 4; ++t) {
      v16bf bf = load_b_frag(Bw, ldb, ncol0 + 16 * t, k0);
      acc[t] = wmma_bf16(af, bf, acc[t]);
    }
  }

#pragma unroll
  for (int t = 0; t < 4; ++t) {
    int col = ncol0 + 16 * t + lidx;
    float bv = bias ? bias[col] : 0.0f;
#pragma unroll
    for (int r = 0; r < 8; ++r) {
      int row = row0 + r + 8 * half;
      float val = acc[t][r] + bv;
      if (MODE == 0) {
        ((u16*)outp)[(size_t)row * N + col] = f2bf(gelu_f(val));
      } else if (MODE == 1) {
        ((u16*)outp)[(size_t)row * N + col] = f2bf(val);
      } else if (MODE == 2) {
        int b = row >> 12, r2 = row & 4095;
        int t_ = r2 >> 8, p = r2 & 255;
        int s = p * TT + t_;
        ((float*)outp)[((size_t)b * NTOK + s) * CB + col] = val;
      } else if (MODE == 3) {
        int b = row >> 12, s = row & 4095;
        ((u16*)outp)[((size_t)b * HID + col) * NTOK + s] = f2bf(val);
      } else if (MODE == 4) {
        size_t idx = (size_t)row * N + col;
        ((float*)outp)[idx] = resid[idx] + val;
      } else if (MODE == 5) {
        int b = row >> 12, s = row & 4095;
        int p = s >> 4, t_ = s & 15;
        size_t oidx = (((size_t)b * CB + col) * TT + t_) * HW + p;
        ((float*)outp)[oidx] = resid[(size_t)row * CB + col] + val;
      }
    }
  }
}

// ---------------------------------------------------------------------------
// Flash attention: 4 waves/block share one (b, head); each wave owns 16 query
// rows.  K/V 32x32 bf16 tiles are staged into LDS by the Tensor Data Mover
// (double-buffered; DMA for block j+1 overlaps compute on block j), so the
// four waves read K/V from LDS instead of 4x-redundant global loads.
// ---------------------------------------------------------------------------
__global__ __launch_bounds__(128) void attn_kernel(const u16* __restrict__ qb,
                                                   const u16* __restrict__ kb,
                                                   const u16* __restrict__ vT,
                                                   u16* __restrict__ ob) {
  __shared__ __align__(16) u16 kst[2][32][32];   // [buf][j][d]
  __shared__ __align__(16) u16 vst[2][32][32];   // [buf][d][j]
  __shared__ __align__(16) u16 pst[4][16][32];   // per-wave P tile staging
  int lane = threadIdx.x & 31;
  int w = threadIdx.x >> 5;
  int half = lane >> 4, lidx = lane & 15;
  int b = blockIdx.z, head = blockIdx.y;
  int i0 = (blockIdx.x * 4 + w) * 16;

  const u16* qptr  = qb + (size_t)b * NTOK * HID;
  const u16* kbase = kb + (size_t)b * NTOK * HID + head * DH;   // row j stride HID
  const u16* vbase = vT + ((size_t)b * HID + head * DH) * NTOK; // row d stride NTOK

  v16bf qf = load_a_frag(qptr, HID, i0, head * DH);
  v8f acc0 = {}, acc1 = {};
  float mrow[8], lrow[8];
#pragma unroll
  for (int r = 0; r < 8; ++r) { mrow[r] = -1e30f; lrow[r] = 0.0f; }

  // stage a 32-column K/V block into LDS buffer `buf`
  auto stage = [&](int buf, int jblk) {
#if HAVE_TDM
    if (w == 0) {
      tdm_stage_32x32(kbase + (size_t)jblk * HID,
                      (u32)(uintptr_t)&kst[buf][0][0], HID);
      tdm_stage_32x32(vbase + jblk,
                      (u32)(uintptr_t)&vst[buf][0][0], NTOK);
    }
#else
    int r = threadIdx.x >> 2, c = (threadIdx.x & 3) * 8;
    *(uint4*)&kst[buf][r][c] = *(const uint4*)(kbase + (size_t)(jblk + r) * HID + c);
    *(uint4*)&vst[buf][r][c] = *(const uint4*)(vbase + (size_t)r * NTOK + jblk + c);
#endif
  };
  auto stage_wait = [&]() {
#if HAVE_TDM
    if (w == 0) __builtin_amdgcn_s_wait_tensorcnt(0);
#endif
    __syncthreads();   // drains DScnt + publishes TDM-written LDS to all waves
  };

  int cur = 0;
  stage(cur, 0);
  stage_wait();

  for (int j0 = 0; j0 < NTOK; j0 += 32) {
    int nxt = cur ^ 1;
    if (j0 + 32 < NTOK) stage(nxt, j0 + 32);   // overlap DMA with compute

    v16bf kf0 = load_b_frag(&kst[cur][0][0], 32, 0, 0);
    v16bf kf1 = load_b_frag(&kst[cur][0][0], 32, 16, 0);
    v8f z = {};
    v8f s0 = wmma_bf16(qf, kf0, z);
    v8f s1 = wmma_bf16(qf, kf1, z);

#pragma unroll
    for (int r = 0; r < 8; ++r) {
      float a0 = s0[r] * ATT_SCALE;
      float a1 = s1[r] * ATT_SCALE;
      float mx = red16_max(fmaxf(a0, a1));
      float mnew = fmaxf(mrow[r], mx);
      float corr = __expf(mrow[r] - mnew);
      float p0 = __expf(a0 - mnew);
      float p1 = __expf(a1 - mnew);
      float ps = red16_sum(p0 + p1);
      lrow[r] = lrow[r] * corr + ps;
      mrow[r] = mnew;
      acc0[r] *= corr;
      acc1[r] *= corr;
      pst[w][r + 8 * half][lidx]      = f2bf(p0);
      pst[w][r + 8 * half][lidx + 16] = f2bf(p1);
    }
    // P: C-layout -> A-fragment layout via wave-private LDS (DS ops in-order)
    FragU pf;
    const u16* sp = &pst[w][lidx][0];
    pf.q[0] = *(const uint4*)(sp + 8 * half);
    pf.q[1] = *(const uint4*)(sp + 16 + 8 * half);

    v16bf vf0 = load_b_frag(&vst[cur][0][0], 32, 0, 0);
    v16bf vf1 = load_b_frag(&vst[cur][0][0], 32, 16, 0);
    acc0 = wmma_bf16(pf.v, vf0, acc0);
    acc1 = wmma_bf16(pf.v, vf1, acc1);

    stage_wait();   // next buffer ready; all waves done reading cur
    cur = nxt;
  }

#pragma unroll
  for (int r = 0; r < 8; ++r) {
    float inv = 1.0f / lrow[r];
    size_t base = ((size_t)b * NTOK + i0 + r + 8 * half) * HID + head * DH;
    ob[base + lidx]      = f2bf(acc0[r] * inv);
    ob[base + lidx + 16] = f2bf(acc1[r] * inv);
  }
}

// ---------------------------------------------------------------------------
// Host launcher
// ---------------------------------------------------------------------------
extern "C" void kernel_launch(void* const* d_in, const int* in_sizes, int n_in,
                              void* d_out, int out_size, void* d_ws, size_t ws_size,
                              hipStream_t stream) {
  const float* x         = (const float*)d_in[0];
  const float* frame_idx = (const float*)d_in[1];
  const float* fusion_w1 = (const float*)d_in[2];
  const float* fusion_b1 = (const float*)d_in[3];
  const float* fusion_w2 = (const float*)d_in[4];
  const float* fusion_b2 = (const float*)d_in[5];
  const float* ln1_g     = (const float*)d_in[6];
  const float* ln1_b     = (const float*)d_in[7];
  const float* wq        = (const float*)d_in[8];
  const float* wk        = (const float*)d_in[9];
  const float* wv        = (const float*)d_in[10];
  const float* wo        = (const float*)d_in[11];
  const float* bo        = (const float*)d_in[12];
  const float* ln2_g     = (const float*)d_in[13];
  const float* ln2_b     = (const float*)d_in[14];
  const float* mlp_w1    = (const float*)d_in[15];
  const float* mlp_b1    = (const float*)d_in[16];
  const float* mlp_w2    = (const float*)d_in[17];
  const float* mlp_b2    = (const float*)d_in[18];
  float* out = (float*)d_out;

  char* base = (char*)d_ws;
  size_t off = 0;
  auto alloc = [&](size_t bytes) -> char* {
    char* p = base + off;
    off += (bytes + 255) & ~(size_t)255;
    return p;
  };
  u16* WT1  = (u16*)alloc((size_t)MLPH * KF * 2);
  u16* WT2  = (u16*)alloc((size_t)CB * MLPH * 2);
  u16* WTQ  = (u16*)alloc((size_t)HID * CB * 2);
  u16* WTK  = (u16*)alloc((size_t)HID * CB * 2);
  u16* WTV  = (u16*)alloc((size_t)HID * CB * 2);
  u16* WTO  = (u16*)alloc((size_t)CB * HID * 2);
  u16* WTM1 = (u16*)alloc((size_t)MLPH * CB * 2);
  u16* WTM2 = (u16*)alloc((size_t)CB * MLPH * 2);
  u16* xf   = (u16*)alloc((size_t)MTOK * KF * 2);
  u16* h1   = (u16*)alloc((size_t)MTOK * MLPH * 2);
  float* xs = (float*)alloc((size_t)MTOK * CB * 4);
  u16* xn   = (u16*)alloc((size_t)MTOK * CB * 2);
  u16* qbuf = (u16*)alloc((size_t)MTOK * HID * 2);
  u16* kbuf = (u16*)alloc((size_t)MTOK * HID * 2);
  u16* vTb  = (u16*)alloc((size_t)MTOK * HID * 2);
  u16* obuf = (u16*)alloc((size_t)MTOK * HID * 2);
  float* xs2 = (float*)alloc((size_t)MTOK * CB * 4);
  u16* xn2  = (u16*)alloc((size_t)MTOK * CB * 2);
  (void)ws_size; (void)in_sizes; (void)n_in; (void)out_size;

  auto gridElems = [](size_t n) { return (unsigned)((n + 255) / 256); };

  // --- weight prep (f32 [K][N] -> bf16 W^T [N][Kp]) ---
  wprep_kernel<<<gridElems((size_t)MLPH * KF), 256, 0, stream>>>(fusion_w1, WT1, CB + 1, MLPH, KF, MLPH * KF);
  wprep_kernel<<<gridElems((size_t)CB * MLPH), 256, 0, stream>>>(fusion_w2, WT2, MLPH, CB, MLPH, CB * MLPH);
  wprep_kernel<<<gridElems((size_t)HID * CB), 256, 0, stream>>>(wq, WTQ, CB, HID, CB, HID * CB);
  wprep_kernel<<<gridElems((size_t)HID * CB), 256, 0, stream>>>(wk, WTK, CB, HID, CB, HID * CB);
  wprep_kernel<<<gridElems((size_t)HID * CB), 256, 0, stream>>>(wv, WTV, CB, HID, CB, HID * CB);
  wprep_kernel<<<gridElems((size_t)CB * HID), 256, 0, stream>>>(wo, WTO, HID, CB, HID, CB * HID);
  wprep_kernel<<<gridElems((size_t)MLPH * CB), 256, 0, stream>>>(mlp_w1, WTM1, CB, MLPH, CB, MLPH * CB);
  wprep_kernel<<<gridElems((size_t)CB * MLPH), 256, 0, stream>>>(mlp_w2, WTM2, MLPH, CB, MLPH, CB * MLPH);

  // --- token packing ---
  pack_x_kernel<<<gridElems((size_t)MTOK * KF), 256, 0, stream>>>(x, frame_idx, xf, MTOK * KF);

  // --- fusion MLP ---
  gemm_kernel<0><<<dim3(MLPH / 128, MTOK / 64), 256, 0, stream>>>(xf, KF, WT1, KF, fusion_b1, MTOK, MLPH, KF, nullptr, h1);
  gemm_kernel<2><<<dim3(CB / 128, MTOK / 64), 256, 0, stream>>>(h1, MLPH, WT2, MLPH, fusion_b2, MTOK, CB, MLPH, nullptr, xs);

  // --- attention ---
  ln_kernel<<<MTOK / 8, 256, 0, stream>>>(xs, ln1_g, ln1_b, xn, MTOK);
  gemm_kernel<1><<<dim3(HID / 128, MTOK / 64), 256, 0, stream>>>(xn, CB, WTQ, CB, nullptr, MTOK, HID, CB, nullptr, qbuf);
  gemm_kernel<1><<<dim3(HID / 128, MTOK / 64), 256, 0, stream>>>(xn, CB, WTK, CB, nullptr, MTOK, HID, CB, nullptr, kbuf);
  gemm_kernel<3><<<dim3(HID / 128, MTOK / 64), 256, 0, stream>>>(xn, CB, WTV, CB, nullptr, MTOK, HID, CB, nullptr, vTb);
  attn_kernel<<<dim3(NTOK / 64, HEADS, 2), 128, 0, stream>>>(qbuf, kbuf, vTb, obuf);
  gemm_kernel<4><<<dim3(CB / 128, MTOK / 64), 256, 0, stream>>>(obuf, HID, WTO, HID, bo, MTOK, CB, HID, xs, xs2);

  // --- post MLP + output transpose ---
  ln_kernel<<<MTOK / 8, 256, 0, stream>>>(xs2, ln2_g, ln2_b, xn2, MTOK);
  gemm_kernel<0><<<dim3(MLPH / 128, MTOK / 64), 256, 0, stream>>>(xn2, CB, WTM1, CB, mlp_b1, MTOK, MLPH, CB, nullptr, h1);
  gemm_kernel<5><<<dim3(CB / 128, MTOK / 64), 256, 0, stream>>>(h1, MLPH, WTM2, MLPH, mlp_b2, MTOK, CB, MLPH, xs2, out);
}
